// Repr2Classifier_15960098472336
// MI455X (gfx1250) — matrix operand order repaired and checked by reading, not compiled, not run on hardware
//
#include <hip/hip_runtime.h>
#include <stdint.h>

// ---------------------------------------------------------------------------
// Problem constants (match reference)
// ---------------------------------------------------------------------------
#define NF    200000
#define NH    20000
#define NEHF  400000
#define NEFF  600000
#define NB    256
#define HDIM  128
#define D0    84
#define D0P   96      // D0 padded to multiple of 32 for WMMA K

#define GEMM_INIT 1
#define GEMM_RELU 2

typedef __attribute__((ext_vector_type(16))) __bf16 v16bf;
typedef __attribute__((ext_vector_type(8)))  float  v8f;

static __device__ __forceinline__ __bf16 to_bf(float x) { return (__bf16)x; }

// ---------------------------------------------------------------------------
// Weight prep: fp32 W[K x N] (optionally + W1) -> bf16 packed in the CDNA5
// WMMA B-fragment slots, in this file's contiguous k order. dst index: ((kt*NT + nt)*32 + lane)*16 + j where
// n = nt*16 + (lane&15), k = kt*32 + (lane>>4)*16 + j; k >= K zero-padded.
// ---------------------------------------------------------------------------
__global__ void prep_weight(const float* __restrict__ W0,
                            const float* __restrict__ W1,
                            __bf16* __restrict__ dst, int K, int N, int ktiles)
{
    int ntiles = N >> 4;
    int total  = ktiles * ntiles * 512;
    int t = blockIdx.x * 256 + threadIdx.x;
    if (t >= total) return;
    int j    = t & 15;
    int lane = (t >> 4) & 31;
    int tile = t >> 9;
    int nt   = tile % ntiles;
    int kt   = tile / ntiles;
    int n = nt * 16 + (lane & 15);
    int k = kt * 32 + (lane >> 4) * 16 + j;
    float v = 0.0f;
    if (k < K) {
        v = W0[(size_t)k * N + n];
        if (W1) v += W1[(size_t)k * N + n];
    }
    dst[t] = (__bf16)v;
}

// ---------------------------------------------------------------------------
// WMMA GEMM: C[M x NT*16] (+)= A[M x lda(fp32)] @ Bpacked (+ bias) (relu).
// M = mtiles*16 (all M here are multiples of 16). One wave per 16-row strip.
// KT is a compile-time k-tile count so the whole MAC pipeline unrolls and the
// scheduler can clause global A loads / LDS B loads across k-tiles.
// ---------------------------------------------------------------------------
template <int NT, int KT>
__global__ __launch_bounds__(256) void gemm_wmma(
    const float* __restrict__ A, int lda,
    const __bf16* __restrict__ Bp,
    const float* __restrict__ bias,
    float* __restrict__ C, int mtiles, int flags)
{
    __shared__ __align__(32) __bf16 sB[KT * NT * 512];   // <= 32 KB
    const int tid = threadIdx.x;
    for (int i = tid; i < KT * NT * 512; i += 256) sB[i] = Bp[i];
    __syncthreads();

    const int lane = tid & 31;
    const int wave = tid >> 5;
    const int l2   = lane & 15;
    const int hi   = lane >> 4;          // A: K half select; C: row-half select
    const int LDC  = NT * 16;

    const v16bf* bfrag = (const v16bf*)sB;

    for (int ws = blockIdx.x * 8 + wave; ws < mtiles; ws += gridDim.x * 8) {
        const int arow = ws * 16 + l2;   // A row held by this lane
        v8f acc[NT];
        const v8f vzero = {0.f,0.f,0.f,0.f,0.f,0.f,0.f,0.f};
#pragma unroll
        for (int nt = 0; nt < NT; ++nt) acc[nt] = vzero;

#pragma unroll
        for (int kt = 0; kt < KT; ++kt) {
            // A fragment: lane holds row arow, K = kt*32 + hi*16 + {0..15}
            const float4* ap = (const float4*)(A + (size_t)arow * lda + kt * 32 + hi * 16);
            float4 a0 = ap[0], a1 = ap[1], a2 = ap[2], a3 = ap[3];
            v16bf af;
            af[0]=to_bf(a0.x); af[1]=to_bf(a0.y); af[2]=to_bf(a0.z); af[3]=to_bf(a0.w);
            af[4]=to_bf(a1.x); af[5]=to_bf(a1.y); af[6]=to_bf(a1.z); af[7]=to_bf(a1.w);
            af[8]=to_bf(a2.x); af[9]=to_bf(a2.y); af[10]=to_bf(a2.z); af[11]=to_bf(a2.w);
            af[12]=to_bf(a3.x); af[13]=to_bf(a3.y); af[14]=to_bf(a3.z); af[15]=to_bf(a3.w);

            // Batch all B-fragment LDS loads first so they issue as one clause
            // and drain with partial s_wait_dscnt instead of 8 full waits.
            v16bf bfr[NT];
#pragma unroll
            for (int nt = 0; nt < NT; ++nt)
                bfr[nt] = bfrag[(kt * NT + nt) * 32 + lane];
#pragma unroll
            for (int nt = 0; nt < NT; ++nt)
                acc[nt] = __builtin_amdgcn_wmma_f32_16x16x32_bf16(
                    false, af, false, bfr[nt], (short)0, acc[nt], false, false);
        }

        // Epilogue: acc[nt] elem r -> row = ws*16 + hi*8 + r, col = nt*16 + l2
#pragma unroll
        for (int nt = 0; nt < NT; ++nt) {
            const int col = nt * 16 + l2;
            float badd = bias ? bias[col] : 0.0f;
#pragma unroll
            for (int r = 0; r < 8; ++r) {
                const int row = ws * 16 + hi * 8 + r;
                size_t idx = (size_t)row * LDC + col;
                float v = acc[nt][r] + badd;
                if (!(flags & GEMM_INIT)) v += C[idx];
                if (flags & GEMM_RELU)    v = fmaxf(v, 0.0f);
                C[idx] = v;
            }
        }
    }
}

// ---------------------------------------------------------------------------
// Edge scatter-add: one wave32 per edge, atomic fp32 add into agg rows.
// ---------------------------------------------------------------------------
__global__ __launch_bounds__(256) void scatter_add(
    const float* __restrict__ X, int ldx,
    const int* __restrict__ src, const int* __restrict__ dst,
    int nE, int dim, float* __restrict__ AGG, int ldagg)
{
    int e = blockIdx.x * 8 + (threadIdx.x >> 5);
    int lane = threadIdx.x & 31;
    if (e >= nE) return;
    const float* xs = X   + (size_t)src[e] * ldx;
    float*       ag = AGG + (size_t)dst[e] * ldagg;
    for (int c = lane; c < dim; c += 32)
        atomicAdd(&ag[c], xs[c]);
}

// ---------------------------------------------------------------------------
// Flow input embedding concat -> xf0 [NF x D0P], cols 84..95 zero.
// ---------------------------------------------------------------------------
__global__ void embed_concat(
    const float* __restrict__ raw, const int* __restrict__ ports,
    const int* __restrict__ fl,  const int* __restrict__ flr,
    const float* __restrict__ pe, const float* __restrict__ fe,
    const float* __restrict__ fre, float* __restrict__ xf0)
{
    long i = (long)blockIdx.x * blockDim.x + threadIdx.x;
    if (i >= (long)NF * D0P) return;
    int f = (int)(i / D0P), c = (int)(i % D0P);
    float v;
    if      (c < 64) v = raw[(size_t)f * 64 + c];
    else if (c < 80) v = pe[(size_t)ports[f] * 16 + (c - 64)];
    else if (c < 82) v = fe[fl[f]  * 2 + (c - 80)];
    else if (c < 84) v = fre[flr[f] * 2 + (c - 82)];
    else             v = 0.0f;
    xf0[i] = v;
}

// ---------------------------------------------------------------------------
// Segment-max pooling via order-preserving uint keys + atomicMax.
// key(f) monotonic in f; key==0 < key(any real) (init by memset 0).
// ---------------------------------------------------------------------------
static __device__ __forceinline__ unsigned f2key(float f) {
    unsigned u = __float_as_uint(f);
    return (u & 0x80000000u) ? ~u : (u | 0x80000000u);
}

__global__ void pool_max(const float* __restrict__ xf,
                         const int* __restrict__ batch,
                         unsigned* __restrict__ keys)
{
    long i = (long)blockIdx.x * blockDim.x + threadIdx.x;
    if (i >= (long)NF * HDIM) return;
    int f = (int)(i >> 7), c = (int)(i & 127);
    atomicMax(&keys[(size_t)batch[f] * HDIM + c], f2key(xf[i]));
}

__global__ void pool_decode(const unsigned* __restrict__ keys,
                            float* __restrict__ out)
{
    int i = blockIdx.x * blockDim.x + threadIdx.x;
    if (i >= NB * HDIM) return;
    unsigned k = keys[i];
    out[i] = (k & 0x80000000u) ? __uint_as_float(k & 0x7fffffffu)
                               : __uint_as_float(~k);
}

// ---------------------------------------------------------------------------
// Host launcher
// ---------------------------------------------------------------------------
extern "C" void kernel_launch(void* const* d_in, const int* in_sizes, int n_in,
                              void* d_out, int out_size, void* d_ws, size_t ws_size,
                              hipStream_t stream)
{
    (void)in_sizes; (void)n_in; (void)out_size; (void)ws_size;
    const float* x_flow_raw   = (const float*)d_in[0];
    const float* x_host       = (const float*)d_in[1];
    const int*   dst_ports    = (const int*)d_in[2];
    const int*   tcp_flags    = (const int*)d_in[3];
    const int*   tcp_flags_rv = (const int*)d_in[4];
    const int*   e_h2f        = (const int*)d_in[5];   // [0:E)=src(host), [E:2E)=dst(flow)
    const int*   e_f2h        = (const int*)d_in[6];
    const int*   e_f2f        = (const int*)d_in[7];
    const int*   batch        = (const int*)d_in[8];
    const float* port_emb     = (const float*)d_in[9];
    const float* flags_emb    = (const float*)d_in[10];
    const float* flags_rv_emb = (const float*)d_in[11];
    const float* W_rel_h2f0   = (const float*)d_in[12];
    const float* b_rel_h2f0   = (const float*)d_in[13];
    const float* W_root_h2f0  = (const float*)d_in[14];
    const float* W_rel_f2h0   = (const float*)d_in[15];
    const float* b_rel_f2h0   = (const float*)d_in[16];
    const float* W_root_f2h0  = (const float*)d_in[17];
    const float* W_rel_f2f0   = (const float*)d_in[18];
    const float* b_rel_f2f0   = (const float*)d_in[19];
    const float* W_root_f2f0  = (const float*)d_in[20];
    const float* Wr_h2f       = (const float*)d_in[21];
    const float* br_h2f       = (const float*)d_in[22];
    const float* Wt_h2f       = (const float*)d_in[23];
    const float* Wr_f2h       = (const float*)d_in[24];
    const float* br_f2h       = (const float*)d_in[25];
    const float* Wt_f2h       = (const float*)d_in[26];
    const float* Wr_f2f       = (const float*)d_in[27];
    const float* br_f2f       = (const float*)d_in[28];
    const float* Wt_f2f       = (const float*)d_in[29];
    const float* W1 = (const float*)d_in[30];
    const float* b1 = (const float*)d_in[31];
    const float* W2 = (const float*)d_in[32];
    const float* b2 = (const float*)d_in[33];
    const float* W3 = (const float*)d_in[34];
    const float* b3 = (const float*)d_in[35];

    // ---- workspace carve -------------------------------------------------
    char*  base = (char*)d_ws;
    size_t off  = 0;
    auto alloc = [&](size_t bytes) -> void* {
        void* p = base + off;
        off = (off + bytes + 255) & ~(size_t)255;
        return p;
    };
    float* xf0   = (float*)alloc((size_t)NF * D0P  * 4);
    float* xfA   = (float*)alloc((size_t)NF * HDIM * 4);
    float* xfB   = (float*)alloc((size_t)NF * HDIM * 4);
    float* agg1  = (float*)alloc((size_t)NF * HDIM * 4);  // h2f aggregation (always 128-d)
    float* agg2  = (float*)alloc((size_t)NF * HDIM * 4);  // f2f aggregation
    float* xhA   = (float*)alloc((size_t)NH * HDIM * 4);
    float* xhB   = (float*)alloc((size_t)NH * HDIM * 4);
    float* aggH  = (float*)alloc((size_t)NH * HDIM * 4);  // f2h aggregation
    unsigned* poolK = (unsigned*)alloc((size_t)NB * HDIM * 4);
    float* poolF = (float*)alloc((size_t)NB * HDIM * 4);
    float* h1    = (float*)alloc((size_t)NB * HDIM * 4);
    float* h2    = (float*)alloc((size_t)NB * HDIM * 4);
    __bf16* P[18];
    for (int i = 0; i < 18; ++i) P[i] = (__bf16*)alloc((size_t)128 * 128 * 2);

    // ---- weight prep (bf16, WMMA-packed) ---------------------------------
    auto prep = [&](__bf16* dst, const float* w0, const float* w1, int K, int N) {
        int ktiles = (K + 31) / 32;
        int total  = ktiles * (N / 16) * 512;
        prep_weight<<<(total + 255) / 256, 256, 0, stream>>>(w0, w1, dst, K, N, ktiles);
    };
    const int HH = HDIM * HDIM;
    prep(P[0], W_rel_h2f0,  nullptr,      HDIM, HDIM);
    prep(P[1], W_root_h2f0, W_root_f2f0,  D0,   HDIM);   // fused flow roots, layer 0
    prep(P[2], W_rel_f2f0,  nullptr,      D0,   HDIM);
    prep(P[3], W_rel_f2h0,  nullptr,      D0,   HDIM);
    prep(P[4], W_root_f2h0, nullptr,      HDIM, HDIM);
    for (int l = 0; l < 2; ++l) {
        prep(P[5 + 5*l], Wr_h2f + l*HH, nullptr,        HDIM, HDIM);
        prep(P[6 + 5*l], Wt_h2f + l*HH, Wt_f2f + l*HH,  HDIM, HDIM); // fused flow roots
        prep(P[7 + 5*l], Wr_f2f + l*HH, nullptr,        HDIM, HDIM);
        prep(P[8 + 5*l], Wr_f2h + l*HH, nullptr,        HDIM, HDIM);
        prep(P[9 + 5*l], Wt_f2h + l*HH, nullptr,        HDIM, HDIM);
    }
    prep(P[15], W1, nullptr, HDIM, HDIM);
    prep(P[16], W2, nullptr, HDIM, HDIM);
    prep(P[17], W3, nullptr, HDIM, 16);

    auto gemm8 = [&](const float* A, int lda, int kt, const __bf16* Bp,
                     const float* bias, float* C, int mtiles, int flags) {
        int blocks = (mtiles + 7) / 8;
        if (kt == 4)
            gemm_wmma<8, 4><<<blocks, 256, 0, stream>>>(A, lda, Bp, bias, C, mtiles, flags);
        else
            gemm_wmma<8, 3><<<blocks, 256, 0, stream>>>(A, lda, Bp, bias, C, mtiles, flags);
    };

    // ---- flow input embedding --------------------------------------------
    {
        long tot = (long)NF * D0P;
        embed_concat<<<(int)((tot + 255) / 256), 256, 0, stream>>>(
            x_flow_raw, dst_ports, tcp_flags, tcp_flags_rv,
            port_emb, flags_emb, flags_rv_emb, xf0);
    }

    // ---- layer 0 (heterogeneous dims: flow side D0->H) -------------------
    hipMemsetAsync(agg1, 0, (size_t)NF * HDIM * 4, stream);
    hipMemsetAsync(agg2, 0, (size_t)NF * D0P  * 4, stream);
    hipMemsetAsync(aggH, 0, (size_t)NH * D0P  * 4, stream);
    scatter_add<<<(NEHF + 7) / 8, 256, 0, stream>>>(x_host, HDIM, e_h2f, e_h2f + NEHF, NEHF, HDIM, agg1, HDIM);
    scatter_add<<<(NEFF + 7) / 8, 256, 0, stream>>>(xf0,    D0P,  e_f2f, e_f2f + NEFF, NEFF, D0,   agg2, D0P);
    scatter_add<<<(NEHF + 7) / 8, 256, 0, stream>>>(xf0,    D0P,  e_f2h, e_f2h + NEHF, NEHF, D0,   aggH, D0P);
    gemm8(agg1,   HDIM, 4, P[0], b_rel_h2f0, xfA, NF / 16, GEMM_INIT);
    gemm8(xf0,    D0P,  3, P[1], nullptr,    xfA, NF / 16, 0);
    gemm8(agg2,   D0P,  3, P[2], b_rel_f2f0, xfA, NF / 16, GEMM_RELU);
    gemm8(aggH,   D0P,  3, P[3], b_rel_f2h0, xhA, NH / 16, GEMM_INIT);
    gemm8(x_host, HDIM, 4, P[4], nullptr,    xhA, NH / 16, GEMM_RELU);

    // ---- layers 1..2 (H->H), relu only after layer 1 ---------------------
    float* xf_cur = xfA; float* xf_nxt = xfB;
    float* xh_cur = xhA; float* xh_nxt = xhB;
    for (int l = 0; l < 2; ++l) {
        const int rl = (l == 0) ? GEMM_RELU : 0;
        hipMemsetAsync(agg1, 0, (size_t)NF * HDIM * 4, stream);
        hipMemsetAsync(agg2, 0, (size_t)NF * HDIM * 4, stream);
        hipMemsetAsync(aggH, 0, (size_t)NH * HDIM * 4, stream);
        scatter_add<<<(NEHF + 7) / 8, 256, 0, stream>>>(xh_cur, HDIM, e_h2f, e_h2f + NEHF, NEHF, HDIM, agg1, HDIM);
        scatter_add<<<(NEFF + 7) / 8, 256, 0, stream>>>(xf_cur, HDIM, e_f2f, e_f2f + NEFF, NEFF, HDIM, agg2, HDIM);
        scatter_add<<<(NEHF + 7) / 8, 256, 0, stream>>>(xf_cur, HDIM, e_f2h, e_f2h + NEHF, NEHF, HDIM, aggH, HDIM);
        gemm8(agg1,   HDIM, 4, P[5 + 5*l], br_h2f + l*HDIM, xf_nxt, NF / 16, GEMM_INIT);
        gemm8(xf_cur, HDIM, 4, P[6 + 5*l], nullptr,         xf_nxt, NF / 16, 0);
        gemm8(agg2,   HDIM, 4, P[7 + 5*l], br_f2f + l*HDIM, xf_nxt, NF / 16, rl);
        gemm8(aggH,   HDIM, 4, P[8 + 5*l], br_f2h + l*HDIM, xh_nxt, NH / 16, GEMM_INIT);
        gemm8(xh_cur, HDIM, 4, P[9 + 5*l], nullptr,         xh_nxt, NH / 16, rl);
        float* t;
        t = xf_cur; xf_cur = xf_nxt; xf_nxt = t;
        t = xh_cur; xh_cur = xh_nxt; xh_nxt = t;
    }

    // ---- global max pool + MLP head --------------------------------------
    hipMemsetAsync(poolK, 0, (size_t)NB * HDIM * 4, stream);
    {
        long tot = (long)NF * HDIM;
        pool_max<<<(int)((tot + 255) / 256), 256, 0, stream>>>(xf_cur, batch, poolK);
    }
    pool_decode<<<(NB * HDIM + 255) / 256, 256, 0, stream>>>(poolK, poolF);

    gemm8(poolF, HDIM, 4, P[15], b1, h1, NB / 16, GEMM_INIT | GEMM_RELU);
    gemm8(h1,    HDIM, 4, P[16], b2, h2, NB / 16, GEMM_INIT | GEMM_RELU);
    gemm_wmma<1, 4><<<2, 256, 0, stream>>>(h2, HDIM, P[17], b3, (float*)d_out, NB / 16, GEMM_INIT);
}
